// MultiScaleAttentionLayer_42305427866159
// MI455X (gfx1250) — matrix operand order repaired and checked by reading, not compiled, not run on hardware
//
#include <hip/hip_runtime.h>
#include <math.h>

#define DEV __device__ __forceinline__

typedef __attribute__((ext_vector_type(8)))  unsigned short v8us;
typedef __attribute__((ext_vector_type(8)))  __bf16         v8bf;
typedef __attribute__((ext_vector_type(16))) __bf16         v16bf;
typedef __attribute__((ext_vector_type(8)))  float          v8f;

static constexpr int B_  = 4;
static constexpr int D_  = 256;
static constexpr int P_  = 128 * 128;  // 16384
static constexpr int NH_ = 8;
static constexpr int DH_ = 32;
static constexpr int DM_ = 1024;

DEV unsigned short f2bf(float f) {
  unsigned int u = __builtin_bit_cast(unsigned int, f);
  unsigned int r = u + 0x7FFFu + ((u >> 16) & 1u);   // round-to-nearest-even
  return (unsigned short)(r >> 16);
}
DEV float bf2f(unsigned short s) {
  unsigned int u = ((unsigned int)s) << 16;
  return __builtin_bit_cast(float, u);
}
DEV float gelu_erf(float x) { return 0.5f * x * (1.0f + erff(x * 0.70710678118654752440f)); }
DEV float phi1(float x) { return x > 0.f ? x + 1.f : expf(x); }  // elu(x)+1
DEV float bn_apply(const float* bn, int C, int c, float x) {
  float g = bn[c], bt = bn[C + c], mu = bn[2 * C + c], var = bn[3 * C + c];
  return (x - mu) * (g * rsqrtf(var + 1e-5f)) + bt;
}

// ---------------- elementwise / layout kernels ----------------

__global__ void k_prenorm(const float* __restrict__ x0, const float* __restrict__ bn,
                          unsigned short* __restrict__ xn, int total) {
  int i = blockIdx.x * 256 + threadIdx.x;
  if (i >= total) return;
  int c = (i >> 14) & (D_ - 1);
  xn[i] = f2bf(gelu_erf(bn_apply(bn, D_, c, x0[i])));
}

__global__ void k_f2bf(const float* __restrict__ in, unsigned short* __restrict__ out, int n) {
  int i = blockIdx.x * 256 + threadIdx.x;
  if (i < n) out[i] = f2bf(in[i]);
}

// scale 1: transpose [b][c][p] -> [b][p][c]
__global__ void k_im2col_s1(const unsigned short* __restrict__ xn,
                            unsigned short* __restrict__ xt, int total) {
  int i = blockIdx.x * 256 + threadIdx.x;
  if (i >= total) return;
  int c = i & (D_ - 1);
  int p = (i >> 8) & (P_ - 1);
  int b = i >> 22;
  xt[i] = xn[((size_t)(b * D_ + c)) * P_ + p];
}

// scale 2: 2x2 stride-2 patches -> [b][p=4096][k=1024], k = cin*4 + ky*2 + kx
__global__ void k_im2col_s2(const unsigned short* __restrict__ xn,
                            unsigned short* __restrict__ xt, int total) {
  int i = blockIdx.x * 256 + threadIdx.x;
  if (i >= total) return;
  int k = i & 1023;
  int p = (i >> 10) & 4095;
  int b = i >> 22;
  int cin = k >> 2, ky = (k >> 1) & 1, kx = k & 1;
  int py = p >> 6, px = p & 63;
  xt[i] = xn[((size_t)(b * D_ + cin)) * P_ + (2 * py + ky) * 128 + 2 * px + kx];
}

// scale 3: 4x4 stride-4 patches -> [b][p=1024][k=4096], k = cin*16 + ky*4 + kx
__global__ void k_im2col_s3(const unsigned short* __restrict__ xn,
                            unsigned short* __restrict__ xt, int total) {
  int i = blockIdx.x * 256 + threadIdx.x;
  if (i >= total) return;
  int k = i & 4095;
  int p = (i >> 12) & 1023;
  int b = i >> 22;
  int cin = k >> 4, ky = (k >> 2) & 3, kx = k & 3;
  int py = p >> 5, px = p & 31;
  xt[i] = xn[((size_t)(b * D_ + cin)) * P_ + (4 * py + ky) * 128 + 4 * px + kx];
}

// ---------------- WMMA GEMM ----------------
// A fragment, 16x32 bf16 (MxK): lane holds row m=lane&15; per lane two contiguous
// 8-element chunks at k = {kb, kb+1..kb+7} and {16+kb ..} with kb = (lane>=16)*8.
DEV v16bf ld_frag_gap16(const unsigned short* p) {
  v8bf lo = __builtin_bit_cast(v8bf, *(const v8us*)p);
  v8bf hi = __builtin_bit_cast(v8bf, *(const v8us*)(p + 16));
  v16bf r;
#pragma unroll
  for (int e = 0; e < 8; ++e) { r[e] = lo[e]; r[e + 8] = hi[e]; }
  return r;
}
// Unused here: k_gemm loads B (column n=lane&15) with ld_frag_gap16, same k order as A;
// this helper reads 16 contiguous K values. X stored [N][K] row-major => contiguous 32B load.
DEV v16bf ld_frag_seq(const unsigned short* p) {
  v8bf lo = __builtin_bit_cast(v8bf, *(const v8us*)p);
  v8bf hi = __builtin_bit_cast(v8bf, *(const v8us*)(p + 8));
  v16bf r;
#pragma unroll
  for (int e = 0; e < 8; ++e) { r[e] = lo[e]; r[e + 8] = hi[e]; }
  return r;
}

// C[b][M][N] = Wt[M,K] * Xt[b][N,K]^T, bf16 in / f32 accumulate.
// MODE 0: Yf = phi(acc)            (Q/K projections)
// MODE 1: Yf = acc                 (V projection)
// MODE 2: v = acc + bias[m] + resid; Yf = v (f32 ch-major) and Yb[b][n][m] = bf16(v)
// MODE 3: Yb[b][m][n] = bf16(gelu(bn(acc)))
// MODE 4: Yf = bn(acc) + resid     (final output)
template <int MODE>
__global__ __launch_bounds__(128) void k_gemm(const unsigned short* __restrict__ Wt,
                                              const unsigned short* __restrict__ Xt,
                                              float* __restrict__ Yf,
                                              unsigned short* __restrict__ Yb,
                                              const float* __restrict__ bn,
                                              const float* __restrict__ bias,
                                              const float* __restrict__ resid,
                                              int M, int N, int K) {
  const int b = blockIdx.z;
  const int lane = threadIdx.x & 31;
  const int wid = threadIdx.x >> 5;
  const int lm = lane & 15;
  const int hi = lane >> 4;
  const int blockM = blockIdx.y * 64 + (wid >> 1) * 32;
  const int blockN = blockIdx.x * 64 + (wid & 1) * 32;

  const unsigned short* Xb = Xt + (size_t)b * (size_t)N * (size_t)K;
  const unsigned short* arow0 = Wt + (size_t)(blockM + lm) * K;
  const unsigned short* arow1 = Wt + (size_t)(blockM + 16 + lm) * K;
  const unsigned short* brow0 = Xb + (size_t)(blockN + lm) * K;
  const unsigned short* brow1 = Xb + (size_t)(blockN + 16 + lm) * K;
  const int kbA = hi * 8, kbB = hi * 8;

  v8f acc[2][2];
#pragma unroll
  for (int i = 0; i < 2; ++i)
#pragma unroll
    for (int j = 0; j < 2; ++j)
#pragma unroll
      for (int t = 0; t < 8; ++t) acc[i][j][t] = 0.f;

#pragma unroll 2
  for (int kt = 0; kt < K; kt += 32) {
    v16bf a0 = ld_frag_gap16(arow0 + kt + kbA);
    v16bf a1 = ld_frag_gap16(arow1 + kt + kbA);
    v16bf b0 = ld_frag_gap16(brow0 + kt + kbB);
    v16bf b1 = ld_frag_gap16(brow1 + kt + kbB);
    __builtin_prefetch(brow0 + kt + 128, 0, 0);   // global_prefetch_b8
    __builtin_prefetch(brow1 + kt + 128, 0, 0);
    acc[0][0] = __builtin_amdgcn_wmma_f32_16x16x32_bf16(false, a0, false, b0, (short)0, acc[0][0], false, false);
    acc[0][1] = __builtin_amdgcn_wmma_f32_16x16x32_bf16(false, a0, false, b1, (short)0, acc[0][1], false, false);
    acc[1][0] = __builtin_amdgcn_wmma_f32_16x16x32_bf16(false, a1, false, b0, (short)0, acc[1][0], false, false);
    acc[1][1] = __builtin_amdgcn_wmma_f32_16x16x32_bf16(false, a1, false, b1, (short)0, acc[1][1], false, false);
  }

  const int hi8 = hi * 8;
#pragma unroll
  for (int i = 0; i < 2; ++i)
#pragma unroll
    for (int j = 0; j < 2; ++j)
#pragma unroll
      for (int r = 0; r < 8; ++r) {
        float v = acc[i][j][r];
        int m = blockM + i * 16 + r + hi8;
        int n = blockN + j * 16 + lm;
        size_t idx = ((size_t)b * M + m) * (size_t)N + n;
        if (MODE == 0) {
          Yf[idx] = phi1(v);
        } else if (MODE == 1) {
          Yf[idx] = v;
        } else if (MODE == 2) {
          v += bias[m] + resid[idx];
          Yf[idx] = v;
          Yb[((size_t)b * N + n) * (size_t)M + m] = f2bf(v);
        } else if (MODE == 3) {
          Yb[idx] = f2bf(gelu_erf(bn_apply(bn, M, m, v)));
        } else {
          Yf[idx] = bn_apply(bn, M, m, v) + resid[idx];
        }
      }
}

// ---------------- linear attention (small reductions, VALU) ----------------
// KV[bh][d][e] = sum_s Kp[bhd][s] * V[bhe][s];  Ks[bh][d] = sum_s Kp
__global__ void k_attn_kv(const float* __restrict__ Kp, const float* __restrict__ V,
                          float* __restrict__ KV, float* __restrict__ Ks, int S) {
  __shared__ float red[256];
  const int tid = threadIdx.x;
  const int bc = blockIdx.x;  // b*256 + h*32 + d
  const int b = bc >> 8;
  const int hd = bc & 255;
  const int h = hd >> 5;
  const int d = hd & 31;
  const float* krow = Kp + (size_t)bc * S;
  const float* vbase = V + ((size_t)(b * D_ + h * DH_)) * S;
  float acc[DH_];
#pragma unroll
  for (int e = 0; e < DH_; ++e) acc[e] = 0.f;
  float ks = 0.f;
  for (int s = tid; s < S; s += 256) {
    float kv = krow[s];
    ks += kv;
    const float* vp = vbase + s;
#pragma unroll
    for (int e = 0; e < DH_; ++e) acc[e] += kv * vp[(size_t)e * S];
  }
  for (int e = 0; e < DH_; ++e) {
    red[tid] = acc[e];
    __syncthreads();
    for (int off = 128; off > 0; off >>= 1) {
      if (tid < off) red[tid] += red[tid + off];
      __syncthreads();
    }
    if (tid == 0) KV[((size_t)((b * NH_ + h) * DH_ + d)) * DH_ + e] = red[0];
    __syncthreads();
  }
  red[tid] = ks;
  __syncthreads();
  for (int off = 128; off > 0; off >>= 1) {
    if (tid < off) red[tid] += red[tid + off];
    __syncthreads();
  }
  if (tid == 0) Ks[(b * NH_ + h) * DH_ + d] = red[0];
}

// O[bhe][p] = (sum_d Qp[bhd][p] * KV[d][e]) / (sum_d Qp*Ks + eps)
__global__ void k_attn_apply(const float* __restrict__ Qp, const float* __restrict__ KV,
                             const float* __restrict__ Ks, float* __restrict__ O, int S) {
  __shared__ float sKV[DH_ * DH_];
  __shared__ float sKs[DH_];
  const int tid = threadIdx.x;
  const int bh = blockIdx.y;  // b*8 + h
  const int b = bh >> 3, h = bh & 7;
  for (int i = tid; i < DH_ * DH_; i += 256) sKV[i] = KV[(size_t)bh * DH_ * DH_ + i];
  if (tid < DH_) sKs[tid] = Ks[bh * DH_ + tid];
  __syncthreads();
  const int p = blockIdx.x * 256 + tid;
  if (p >= S) return;
  const float* qb = Qp + ((size_t)(b * D_ + h * DH_)) * S + p;
  float q[DH_];
#pragma unroll
  for (int d = 0; d < DH_; ++d) q[d] = qb[(size_t)d * S];
  float den = 1e-6f;
#pragma unroll
  for (int d = 0; d < DH_; ++d) den += q[d] * sKs[d];
  float z = 1.0f / den;
  float* ob = O + ((size_t)(b * D_ + h * DH_)) * S + p;
#pragma unroll
  for (int e = 0; e < DH_; ++e) {
    float v = 0.f;
#pragma unroll
    for (int d = 0; d < DH_; ++d) v += q[d] * sKV[d * DH_ + e];
    ob[(size_t)e * S] = v * z;
  }
}

// softmax(scale_weight)-mix of o1 + bilinear-up(o2,2) + bilinear-up(o3,4) (align_corners)
// output: message bf16 pixel-major [b][p][c]
__global__ void k_combine(const float* __restrict__ o1, const float* __restrict__ o2,
                          const float* __restrict__ o3, const float* __restrict__ sw3,
                          unsigned short* __restrict__ msg, int total) {
  int i = blockIdx.x * 256 + threadIdx.x;
  if (i >= total) return;
  int p = i & (P_ - 1);
  int c = (i >> 14) & (D_ - 1);
  int b = i >> 22;
  int y = p >> 7, x = p & 127;
  float s0 = sw3[0], s1 = sw3[1], s2 = sw3[2];
  float mx = fmaxf(s0, fmaxf(s1, s2));
  float e0 = expf(s0 - mx), e1 = expf(s1 - mx), e2 = expf(s2 - mx);
  float inv = 1.f / (e0 + e1 + e2);
  float w0 = e0 * inv, w1 = e1 * inv, w2 = e2 * inv;

  float v1 = o1[((size_t)(b * D_ + c)) * P_ + p];

  const float* b2 = o2 + ((size_t)(b * D_ + c)) * 4096;
  float fy = y * (63.0f / 127.0f); int y0 = (int)fy; float ty = fy - y0;
  int y1 = (y0 + 1 < 64) ? y0 + 1 : 63;
  float fx = x * (63.0f / 127.0f); int x0i = (int)fx; float tx = fx - x0i;
  int x1 = (x0i + 1 < 64) ? x0i + 1 : 63;
  float v2 = (1 - ty) * ((1 - tx) * b2[y0 * 64 + x0i] + tx * b2[y0 * 64 + x1]) +
             ty * ((1 - tx) * b2[y1 * 64 + x0i] + tx * b2[y1 * 64 + x1]);

  const float* b3 = o3 + ((size_t)(b * D_ + c)) * 1024;
  float gy = y * (31.0f / 127.0f); int z0 = (int)gy; float tz = gy - z0;
  int z1 = (z0 + 1 < 32) ? z0 + 1 : 31;
  float gx = x * (31.0f / 127.0f); int u0 = (int)gx; float tu = gx - u0;
  int u1 = (u0 + 1 < 32) ? u0 + 1 : 31;
  float v3 = (1 - tz) * ((1 - tu) * b3[z0 * 32 + u0] + tu * b3[z0 * 32 + u1]) +
             tz * ((1 - tu) * b3[z1 * 32 + u0] + tu * b3[z1 * 32 + u1]);

  msg[((size_t)(b * P_ + p)) * D_ + c] = f2bf(w0 * v1 + w1 * v2 + w2 * v3);
}

// 3x3 depthwise conv (pad 1) + bn_mb2 + gelu; bf16 ch-major in -> bf16 pixel-major out
__global__ void k_dwconv(const unsigned short* __restrict__ hm1, const float* __restrict__ wdw,
                         const float* __restrict__ bn2, unsigned short* __restrict__ hm2,
                         int total) {
  int i = blockIdx.x * 256 + threadIdx.x;
  if (i >= total) return;
  int p = i & (P_ - 1);
  int ch = (i >> 14) & (DM_ - 1);
  int b = i >> 24;
  int y = p >> 7, x = p & 127;
  const unsigned short* base = hm1 + ((size_t)(b * DM_ + ch)) * P_;
  const float* w = wdw + ch * 9;
  float acc = 0.f;
#pragma unroll
  for (int ky = 0; ky < 3; ++ky) {
    int yy = y + ky - 1;
    if ((unsigned)yy >= 128u) continue;
#pragma unroll
    for (int kx = 0; kx < 3; ++kx) {
      int xx = x + kx - 1;
      if ((unsigned)xx >= 128u) continue;
      acc += bf2f(base[yy * 128 + xx]) * w[ky * 3 + kx];
    }
  }
  float v = gelu_erf(bn_apply(bn2, DM_, ch, acc));
  hm2[((size_t)(b * P_ + p)) * DM_ + ch] = f2bf(v);
}

// ---------------- host ----------------
extern "C" void kernel_launch(void* const* d_in, const int* in_sizes, int n_in,
                              void* d_out, int out_size, void* d_ws, size_t ws_size,
                              hipStream_t stream) {
  (void)in_sizes; (void)n_in; (void)out_size; (void)ws_size;
  const float* x0 = (const float*)d_in[0];
  const float* sw = (const float*)d_in[1];
  const float* bn_pre = (const float*)d_in[2];
  const float* wq[3] = {(const float*)d_in[3], (const float*)d_in[6], (const float*)d_in[9]};
  const float* wk[3] = {(const float*)d_in[4], (const float*)d_in[7], (const float*)d_in[10]};
  const float* wv[3] = {(const float*)d_in[5], (const float*)d_in[8], (const float*)d_in[11]};
  const float* merge_w = (const float*)d_in[12];
  const float* merge_b = (const float*)d_in[13];
  const float* mb_w1 = (const float*)d_in[14];
  const float* bn_mb1 = (const float*)d_in[15];
  const float* mb_wdw = (const float*)d_in[16];
  const float* bn_mb2 = (const float*)d_in[17];
  const float* mb_w2 = (const float*)d_in[18];
  const float* bn_mb3 = (const float*)d_in[19];

  char* ws = (char*)d_ws;
  size_t cur = 0;
  auto take = [&](size_t bytes) -> char* {
    char* pp = ws + cur;
    cur += (bytes + 255) & ~(size_t)255;
    return pp;
  };

  unsigned short* xn = (unsigned short*)take((size_t)B_ * D_ * P_ * 2);
  unsigned short* xt1 = (unsigned short*)take((size_t)B_ * P_ * D_ * 2);
  unsigned short* xt2 = (unsigned short*)take((size_t)B_ * 4096 * 1024 * 2);
  unsigned short* xt3 = (unsigned short*)take((size_t)B_ * 1024 * 4096 * 2);
  const int wsz[3] = {D_ * D_, D_ * D_ * 4, D_ * D_ * 16};
  unsigned short *wqb[3], *wkb[3], *wvb[3];
  for (int s = 0; s < 3; ++s) {
    wqb[s] = (unsigned short*)take((size_t)wsz[s] * 2);
    wkb[s] = (unsigned short*)take((size_t)wsz[s] * 2);
    wvb[s] = (unsigned short*)take((size_t)wsz[s] * 2);
  }
  unsigned short* mwb = (unsigned short*)take((size_t)D_ * D_ * 2);
  unsigned short* w1b = (unsigned short*)take((size_t)DM_ * D_ * 2);
  unsigned short* w2b = (unsigned short*)take((size_t)D_ * DM_ * 2);
  const int Ssz[3] = {P_, 4096, 1024};
  const int Kdim[3] = {256, 1024, 4096};
  float *Q[3], *Kk[3], *Vv[3], *Oo[3], *KVs[3], *Kss[3];
  for (int s = 0; s < 3; ++s) {
    Q[s] = (float*)take((size_t)B_ * D_ * Ssz[s] * 4);
    Kk[s] = (float*)take((size_t)B_ * D_ * Ssz[s] * 4);
    Vv[s] = (float*)take((size_t)B_ * D_ * Ssz[s] * 4);
    Oo[s] = (float*)take((size_t)B_ * D_ * Ssz[s] * 4);
    KVs[s] = (float*)take((size_t)B_ * NH_ * DH_ * DH_ * 4);
    Kss[s] = (float*)take((size_t)B_ * NH_ * DH_ * 4);
  }
  unsigned short* msg = (unsigned short*)take((size_t)B_ * P_ * D_ * 2);
  float* xf = (float*)take((size_t)B_ * D_ * P_ * 4);
  unsigned short* xb = (unsigned short*)take((size_t)B_ * P_ * D_ * 2);
  unsigned short* hm1 = (unsigned short*)take((size_t)B_ * DM_ * P_ * 2);
  unsigned short* hm2 = (unsigned short*)take((size_t)B_ * P_ * DM_ * 2);

  auto g1 = [](long n) { return dim3((unsigned)((n + 255) / 256)); };

  k_prenorm<<<g1((long)B_ * D_ * P_), 256, 0, stream>>>(x0, bn_pre, xn, B_ * D_ * P_);

  for (int s = 0; s < 3; ++s) {
    k_f2bf<<<g1(wsz[s]), 256, 0, stream>>>(wq[s], wqb[s], wsz[s]);
    k_f2bf<<<g1(wsz[s]), 256, 0, stream>>>(wk[s], wkb[s], wsz[s]);
    k_f2bf<<<g1(wsz[s]), 256, 0, stream>>>(wv[s], wvb[s], wsz[s]);
  }
  k_f2bf<<<g1(D_ * D_), 256, 0, stream>>>(merge_w, mwb, D_ * D_);
  k_f2bf<<<g1(DM_ * D_), 256, 0, stream>>>(mb_w1, w1b, DM_ * D_);
  k_f2bf<<<g1(D_ * DM_), 256, 0, stream>>>(mb_w2, w2b, D_ * DM_);

  k_im2col_s1<<<g1((long)B_ * P_ * D_), 256, 0, stream>>>(xn, xt1, B_ * P_ * D_);
  k_im2col_s2<<<g1((long)B_ * 4096 * 1024), 256, 0, stream>>>(xn, xt2, B_ * 4096 * 1024);
  k_im2col_s3<<<g1((long)B_ * 1024 * 4096), 256, 0, stream>>>(xn, xt3, B_ * 1024 * 4096);

  const unsigned short* xts[3] = {xt1, xt2, xt3};
  for (int s = 0; s < 3; ++s) {
    dim3 g(Ssz[s] / 64, D_ / 64, B_);
    k_gemm<0><<<g, 128, 0, stream>>>(wqb[s], xts[s], Q[s], nullptr, nullptr, nullptr, nullptr, D_, Ssz[s], Kdim[s]);
    k_gemm<0><<<g, 128, 0, stream>>>(wkb[s], xts[s], Kk[s], nullptr, nullptr, nullptr, nullptr, D_, Ssz[s], Kdim[s]);
    k_gemm<1><<<g, 128, 0, stream>>>(wvb[s], xts[s], Vv[s], nullptr, nullptr, nullptr, nullptr, D_, Ssz[s], Kdim[s]);
  }

  for (int s = 0; s < 3; ++s) {
    k_attn_kv<<<dim3(B_ * D_), 256, 0, stream>>>(Kk[s], Vv[s], KVs[s], Kss[s], Ssz[s]);
    k_attn_apply<<<dim3(Ssz[s] / 256, B_ * NH_), 256, 0, stream>>>(Q[s], KVs[s], Kss[s], Oo[s], Ssz[s]);
  }

  k_combine<<<g1((long)B_ * D_ * P_), 256, 0, stream>>>(Oo[0], Oo[1], Oo[2], sw, msg, B_ * D_ * P_);

  {
    dim3 g(P_ / 64, D_ / 64, B_);
    k_gemm<2><<<g, 128, 0, stream>>>(mwb, msg, xf, xb, nullptr, merge_b, x0, D_, P_, D_);
  }
  {
    dim3 g(P_ / 64, DM_ / 64, B_);
    k_gemm<3><<<g, 128, 0, stream>>>(w1b, xb, nullptr, hm1, bn_mb1, nullptr, nullptr, DM_, P_, D_);
  }
  k_dwconv<<<g1((long)B_ * DM_ * P_), 256, 0, stream>>>(hm1, mb_wdw, bn_mb2, hm2, B_ * DM_ * P_);
  {
    dim3 g(P_ / 64, D_ / 64, B_);
    k_gemm<4><<<g, 128, 0, stream>>>(w2b, hm2, (float*)d_out, nullptr, bn_mb3, nullptr, xf, D_, P_, DM_);
  }
}
